// LSTM_85572928405890
// MI455X (gfx1250) — compile-verified, matrix operand order checked by reading, not run on hardware
//
#include <hip/hip_runtime.h>
#include <hip/hip_bf16.h>

// ---------------------------------------------------------------------------
// Peephole LSTM for MI455X (gfx1250, wave32, WMMA).
//  Phase 0: convert iterator + all weights to bf16; weights stored transposed
//           [N][K] so WMMA B-fragments are contiguous 16B loads per lane.
//  Phase 1: fused input projections P[t][gate][b][h] = x_t @ Wx + b (one GEMM,
//           M=T*B=32768, K=512, N=4*H=4096) with f32 accumulation.
//  Phase 2: persistent recurrent kernel, 64 blocks x 256 threads, grid-wide
//           atomic barrier; h/c staged into LDS with CDNA5 async-to-LDS copies
//           (ASYNCcnt), fp32 master cell state, v_wmma_f32_16x16x32_bf16.
// ---------------------------------------------------------------------------

#define T_DIM 512
#define B_DIM 64
#define X_DIM 512
#define H_DIM 1024

typedef __bf16 bf16_t;
typedef __attribute__((ext_vector_type(16))) __bf16 v16bf;
typedef __attribute__((ext_vector_type(8)))  float  v8f;
typedef __attribute__((ext_vector_type(4)))  float  f32x4;

union FragU { v16bf v; f32x4 q[2]; };

// Load one 16x32 bf16 WMMA fragment (A from row-major [M][K], or B from
// K-transposed weight storage [N][K]).  Per-lane: row = lane&15,
// K-half = lane>>4; elements e<8 -> K = kh*8+e, e>=8 -> K = 16+kh*8+(e-8).
// => exactly two contiguous 16-byte loads per lane.
__device__ inline v16bf load_frag(const bf16_t* p, int ld) {
  const int lane = threadIdx.x & 31;
  const int r    = lane & 15;
  const int kh   = lane >> 4;
  const bf16_t* rp = p + (size_t)r * ld + kh * 8;
  FragU f;
  f.q[0] = *(const f32x4*)(rp);
  f.q[1] = *(const f32x4*)(rp + 16);
  return f.v;
}

__device__ inline float sigmoidf_(float x) { return 1.0f / (1.0f + __expf(-x)); }

// ---------------------------------------------------------------------------
// CDNA5 async memory->LDS copy (ASYNCcnt tracked, no VGPR data round-trip).
// ---------------------------------------------------------------------------
typedef __attribute__((address_space(3))) char lds_char;

__device__ inline void async_ld_b128(void* ldsdst, const void* gsrc) {
  unsigned loff = (unsigned)(uintptr_t)(lds_char*)ldsdst;  // 32-bit LDS addr
  asm volatile("global_load_async_to_lds_b128 %0, %1, off"
               :: "v"(loff), "v"(gsrc)
               : "memory");
}
__device__ inline void wait_async0() {
  asm volatile("s_wait_asynccnt 0x0" ::: "memory");
}

// ---------------------------------------------------------------------------
// Converters
// ---------------------------------------------------------------------------
__global__ void k_cvt_f32_bf16(const float* __restrict__ src,
                               bf16_t* __restrict__ dst, int n) {
  for (int i = blockIdx.x * blockDim.x + threadIdx.x; i < n;
       i += gridDim.x * blockDim.x)
    dst[i] = (bf16_t)src[i];
}

// Wt[n*K + k] = W[k*N + n]  (transpose + convert; reads coalesced over n)
__global__ void k_cvt_wT(const float* __restrict__ W, bf16_t* __restrict__ Wt,
                         int K, int N) {
  int total = K * N;
  for (int i = blockIdx.x * blockDim.x + threadIdx.x; i < total;
       i += gridDim.x * blockDim.x) {
    int n = i % N;
    int k = i / N;
    Wt[(size_t)n * K + k] = (bf16_t)W[(size_t)k * N + n];
  }
}

// ---------------------------------------------------------------------------
// Phase 1: input projections.  P[((t*4+g)*64+b)*H + h]
// grid = (M/32, N/512); block = 256 (8 waves); wave owns 64 output columns
// across two 16-row M-tiles (each B fragment feeds 2 WMMAs).
// ---------------------------------------------------------------------------
#define PROJ_LDA 520  // 512 + 8 bf16 pad -> 1040B rows, kills LDS bank conflicts

__global__ void __launch_bounds__(256)
k_proj(const bf16_t* __restrict__ Xbf, const bf16_t* __restrict__ Wxt,
       const float* __restrict__ bi, const float* __restrict__ bfg,
       const float* __restrict__ bc, const float* __restrict__ bo,
       float* __restrict__ P) {
  __shared__ bf16_t Alds[32 * PROJ_LDA];

  const int tid  = threadIdx.x;
  const int wave = tid >> 5;
  const int lane = tid & 31;
  const int m0   = blockIdx.x * 32;           // rows of [T*B]
  const int ng0  = blockIdx.y * 512;          // cols of [4H]
  const int n0   = ng0 + wave * 64;

  const int g = ng0 >> 10;                    // gate id, constant per block
  const float* bias = (g == 0) ? bi : (g == 1) ? bfg : (g == 2) ? bc : bo;

  // stage A tile: 32 rows x 512 K (bf16), async 16B chunks -> LDS
  for (int idx = tid; idx < 32 * 64; idx += 256) {
    int row = idx >> 6, ck = (idx & 63) * 8;
    async_ld_b128(&Alds[row * PROJ_LDA + ck],
                  &Xbf[(size_t)(m0 + row) * X_DIM + ck]);
  }
  wait_async0();
  __syncthreads();

  v8f acc[2][4];
#pragma unroll
  for (int mt = 0; mt < 2; ++mt)
#pragma unroll
    for (int j = 0; j < 4; ++j) acc[mt][j] = {};

  for (int k0 = 0; k0 < X_DIM; k0 += 32) {
    v16bf va0 = load_frag(&Alds[k0], PROJ_LDA);
    v16bf va1 = load_frag(&Alds[16 * PROJ_LDA + k0], PROJ_LDA);
#pragma unroll
    for (int j = 0; j < 4; ++j) {
      v16bf vb = load_frag(Wxt + (size_t)(n0 + j * 16) * X_DIM + k0, X_DIM);
      acc[0][j] = __builtin_amdgcn_wmma_f32_16x16x32_bf16(
          false, va0, false, vb, (short)0, acc[0][j], false, false);
      acc[1][j] = __builtin_amdgcn_wmma_f32_16x16x32_bf16(
          false, va1, false, vb, (short)0, acc[1][j], false, false);
    }
  }

  // store: D layout -> element r: M = r + 8*(lane>>4), N = lane&15
#pragma unroll
  for (int mt = 0; mt < 2; ++mt) {
#pragma unroll
    for (int j = 0; j < 4; ++j) {
      int n = n0 + j * 16 + (lane & 15);
      int h = n & (H_DIM - 1);
      float bv = bias[h];
#pragma unroll
      for (int r = 0; r < 8; ++r) {
        int mm = m0 + mt * 16 + r + 8 * (lane >> 4);
        int tt = mm >> 6, bb = mm & 63;
        P[(((size_t)tt * 4 + g) * B_DIM + bb) * H_DIM + h] = acc[mt][j][r] + bv;
      }
    }
  }
}

// ---------------------------------------------------------------------------
// Grid-wide barrier (sense via generation counter in global memory)
// ---------------------------------------------------------------------------
__device__ inline void grid_barrier(unsigned* bar, unsigned nblk) {
  __syncthreads();
  if (threadIdx.x == 0) {
    unsigned g = __hip_atomic_load(&bar[1], __ATOMIC_RELAXED,
                                   __HIP_MEMORY_SCOPE_AGENT);
    unsigned a = __hip_atomic_fetch_add(&bar[0], 1u, __ATOMIC_ACQ_REL,
                                        __HIP_MEMORY_SCOPE_AGENT);
    if (a == nblk - 1) {
      __hip_atomic_store(&bar[0], 0u, __ATOMIC_RELAXED,
                         __HIP_MEMORY_SCOPE_AGENT);
      __hip_atomic_store(&bar[1], g + 1u, __ATOMIC_RELEASE,
                         __HIP_MEMORY_SCOPE_AGENT);
    } else {
      while (__hip_atomic_load(&bar[1], __ATOMIC_ACQUIRE,
                               __HIP_MEMORY_SCOPE_AGENT) == g) {
        __builtin_amdgcn_s_sleep(1);
      }
    }
  }
  __syncthreads();
}

// ---------------------------------------------------------------------------
// Phase 2: persistent recurrent kernel.
//  64 blocks x 256 threads; block owns 16 columns of H.
//  LDS: h (64 x 1024 bf16, padded), c (same), exchange buffers (6x64x16 f32).
//  Phase A per step: wave w -> strip s=w>>1 of {i,f,cbar,o_part}, two row
//  tiles; one B fragment feeds two WMMAs.  elementwise -> c_t; grid barrier;
//  Phase B: pre_o += c_t@Woc -> h_t; barrier.
// ---------------------------------------------------------------------------
#define LDH 1032  // 1024 + 8 bf16 pad

__global__ void __launch_bounds__(256)
k_lstm(const float* __restrict__ P, const bf16_t* __restrict__ Wht,
       const bf16_t* __restrict__ Wct, float* __restrict__ c32,
       bf16_t* __restrict__ hbf, bf16_t* __restrict__ cbf, unsigned* bar,
       float* __restrict__ out) {
  extern __shared__ char smem[];
  bf16_t* h_lds = (bf16_t*)smem;                                   // 64*LDH
  bf16_t* c_lds = (bf16_t*)(smem + (size_t)64 * LDH * 2);          // 64*LDH
  float*  ex    = (float*)(smem + (size_t)2 * 64 * LDH * 2);       // [6][64][16]

  const int tid  = threadIdx.x;
  const int wave = tid >> 5;
  const int lane = tid & 31;
  const int n0   = blockIdx.x * 16;   // this block's H columns
  const unsigned nblk = gridDim.x;

  for (int t = 0; t < T_DIM; ++t) {
    // ---- stage h_{t-1}, c_{t-1} (bf16) into LDS via async copies ----
    for (int idx = tid; idx < 64 * 128; idx += 256) {
      int row = idx >> 7, ck = (idx & 127) * 8;
      async_ld_b128(&h_lds[row * LDH + ck], &hbf[(size_t)row * H_DIM + ck]);
      async_ld_b128(&c_lds[row * LDH + ck], &cbf[(size_t)row * H_DIM + ck]);
    }
    wait_async0();
    __syncthreads();

    // ---- phase A GEMMs: wave w -> strip s = w>>1, row tiles rtb, rtb+1 ----
    {
      int s   = wave >> 1;        // 0:i 1:f 2:cbar 3:o_part
      int rtb = (wave & 1) * 2;   // first of two batch row tiles
      v8f acc0 = {}, acc1 = {};
      const bf16_t* arow0  = h_lds + (size_t)rtb * 16 * LDH;
      const bf16_t* arow1  = arow0 + (size_t)16 * LDH;
      const bf16_t* brow   = Wht + ((size_t)s * H_DIM + n0) * H_DIM;
      const bf16_t* carow0 = c_lds + (size_t)rtb * 16 * LDH;
      const bf16_t* carow1 = carow0 + (size_t)16 * LDH;
      const bf16_t* cbrow  = Wct + ((size_t)s * H_DIM + n0) * H_DIM;
      for (int k0 = 0; k0 < H_DIM; k0 += 32) {
        v16bf vb = load_frag(brow + k0, H_DIM);
        acc0 = __builtin_amdgcn_wmma_f32_16x16x32_bf16(
            false, load_frag(arow0 + k0, LDH), false, vb, (short)0, acc0,
            false, false);
        acc1 = __builtin_amdgcn_wmma_f32_16x16x32_bf16(
            false, load_frag(arow1 + k0, LDH), false, vb, (short)0, acc1,
            false, false);
        if (s < 2) {  // peephole on c_{t-1} for i,f gates (wave-uniform)
          v16bf vcb = load_frag(cbrow + k0, H_DIM);
          acc0 = __builtin_amdgcn_wmma_f32_16x16x32_bf16(
              false, load_frag(carow0 + k0, LDH), false, vcb, (short)0, acc0,
              false, false);
          acc1 = __builtin_amdgcn_wmma_f32_16x16x32_bf16(
              false, load_frag(carow1 + k0, LDH), false, vcb, (short)0, acc1,
              false, false);
        }
      }
      int nn = lane & 15;
      int mh = (lane >> 4) * 8;
#pragma unroll
      for (int r = 0; r < 8; ++r)
        ex[((size_t)s * 64 + rtb * 16 + mh + r) * 16 + nn] = acc0[r];
#pragma unroll
      for (int r = 0; r < 8; ++r)
        ex[((size_t)s * 64 + (rtb + 1) * 16 + mh + r) * 16 + nn] = acc1[r];
    }
    __syncthreads();

    // ---- elementwise A: gates i,f,cbar -> c_t ----
    for (int idx = tid; idx < 64 * 16; idx += 256) {
      int bb = idx >> 4, nn = idx & 15, n = n0 + nn;
      size_t pb = ((size_t)t * 4) * B_DIM;
      float pxi = P[((pb + 0 * B_DIM) + bb) * H_DIM + n];
      float pxf = P[((pb + 1 * B_DIM) + bb) * H_DIM + n];
      float pxc = P[((pb + 2 * B_DIM) + bb) * H_DIM + n];
      float pxo = P[((pb + 3 * B_DIM) + bb) * H_DIM + n];
      float iv = sigmoidf_(ex[(0 * 64 + bb) * 16 + nn] + pxi);
      float fv = sigmoidf_(ex[(1 * 64 + bb) * 16 + nn] + pxf);
      float cb = tanhf(ex[(2 * 64 + bb) * 16 + nn] + pxc);
      float cp = c32[(size_t)bb * H_DIM + n];
      float cn = fv * cp + iv * cb;
      c32[(size_t)bb * H_DIM + n] = cn;          // fp32 master cell state
      cbf[(size_t)bb * H_DIM + n] = (bf16_t)cn;  // bf16 mirror for WMMA
      ex[(4 * 64 + bb) * 16 + nn] = cn;          // stash for phase B
      ex[(3 * 64 + bb) * 16 + nn] += pxo;        // o pre-act (x + h@Woh)
    }
    __threadfence();
    grid_barrier(bar, nblk);

    // ---- phase B: restage full c_t (async), add c_t @ W_oc ----
    for (int idx = tid; idx < 64 * 128; idx += 256) {
      int row = idx >> 7, ck = (idx & 127) * 8;
      async_ld_b128(&c_lds[row * LDH + ck], &cbf[(size_t)row * H_DIM + ck]);
    }
    wait_async0();
    __syncthreads();
    if (wave < 4) {
      int rt = wave;
      v8f acc = {};
      const bf16_t* arow = c_lds + (size_t)rt * 16 * LDH;
      const bf16_t* brow = Wct + ((size_t)2 * H_DIM + n0) * H_DIM;  // W_oc
      for (int k0 = 0; k0 < H_DIM; k0 += 32) {
        v16bf va = load_frag(arow + k0, LDH);
        v16bf vb = load_frag(brow + k0, H_DIM);
        acc = __builtin_amdgcn_wmma_f32_16x16x32_bf16(
            false, va, false, vb, (short)0, acc, false, false);
      }
      int m  = rt * 16 + (lane >> 4) * 8;
      int nn = lane & 15;
#pragma unroll
      for (int r = 0; r < 8; ++r)
        ex[((size_t)5 * 64 + m + r) * 16 + nn] = acc[r];
    }
    __syncthreads();

    // ---- elementwise B: o gate, h_t ----
    for (int idx = tid; idx < 64 * 16; idx += 256) {
      int bb = idx >> 4, nn = idx & 15, n = n0 + nn;
      float po = ex[(3 * 64 + bb) * 16 + nn] + ex[(5 * 64 + bb) * 16 + nn];
      float o = sigmoidf_(po);
      float cn = ex[(4 * 64 + bb) * 16 + nn];
      float hv = o * tanhf(cn);
      hbf[(size_t)bb * H_DIM + n] = (bf16_t)hv;
      if (t == T_DIM - 1) out[(size_t)bb * H_DIM + n] = hv;
    }
    __threadfence();
    grid_barrier(bar, nblk);
  }
}

// ---------------------------------------------------------------------------
// Host launcher
// ---------------------------------------------------------------------------
extern "C" void kernel_launch(void* const* d_in, const int* in_sizes, int n_in,
                              void* d_out, int out_size, void* d_ws,
                              size_t ws_size, hipStream_t stream) {
  (void)in_sizes; (void)n_in; (void)out_size; (void)ws_size;

  const float* it  = (const float*)d_in[0];
  const float* Wix = (const float*)d_in[2];
  const float* Wih = (const float*)d_in[3];
  const float* Wic = (const float*)d_in[4];
  const float* Wfx = (const float*)d_in[5];
  const float* Wfh = (const float*)d_in[6];
  const float* Wfc = (const float*)d_in[7];
  const float* Wcx = (const float*)d_in[8];
  const float* Wch = (const float*)d_in[9];
  const float* Wox = (const float*)d_in[10];
  const float* Woh = (const float*)d_in[11];
  const float* Woc = (const float*)d_in[12];
  const float* bi  = (const float*)d_in[13];
  const float* bfv = (const float*)d_in[14];
  const float* bc  = (const float*)d_in[15];
  const float* bo  = (const float*)d_in[16];

  char* ws = (char*)d_ws;
  // workspace layout (bytes)
  const size_t OFF_C32 = 256;                                  // 64*1024 f32
  const size_t OFF_HBF = OFF_C32 + (size_t)B_DIM * H_DIM * 4;  // bf16
  const size_t OFF_CBF = OFF_HBF + (size_t)B_DIM * H_DIM * 2;
  const size_t OFF_ZEND = OFF_CBF + (size_t)B_DIM * H_DIM * 2; // zeroed region
  const size_t OFF_WHT = OFF_ZEND;                             // 4*H*H bf16
  const size_t OFF_WCT = OFF_WHT + (size_t)4 * H_DIM * H_DIM * 2;  // 3*H*H
  const size_t OFF_WXT = OFF_WCT + (size_t)3 * H_DIM * H_DIM * 2;  // 4*H*X
  const size_t OFF_XBF = OFF_WXT + (size_t)4 * H_DIM * X_DIM * 2;  // T*B*X
  const size_t OFF_P   = OFF_XBF + (size_t)T_DIM * B_DIM * X_DIM * 2;

  float*  C32 = (float*)(ws + OFF_C32);
  bf16_t* Hbf = (bf16_t*)(ws + OFF_HBF);
  bf16_t* Cbf = (bf16_t*)(ws + OFF_CBF);
  bf16_t* Wht = (bf16_t*)(ws + OFF_WHT);
  bf16_t* Wct = (bf16_t*)(ws + OFF_WCT);
  bf16_t* Wxt = (bf16_t*)(ws + OFF_WXT);
  bf16_t* Xbf = (bf16_t*)(ws + OFF_XBF);
  float*  P   = (float*)(ws + OFF_P);

  // zero barrier + h0/c0 state
  hipMemsetAsync(d_ws, 0, OFF_ZEND, stream);

  // convert iterator to bf16
  k_cvt_f32_bf16<<<4096, 256, 0, stream>>>(it, Xbf, T_DIM * B_DIM * X_DIM);

  // transposed bf16 weights: Wxt gate order i,f,c,o ; Wht i,f,c,o ; Wct ic,fc,oc
  const size_t HX = (size_t)H_DIM * X_DIM, HH = (size_t)H_DIM * H_DIM;
  k_cvt_wT<<<512, 256, 0, stream>>>(Wix, Wxt + 0 * HX, X_DIM, H_DIM);
  k_cvt_wT<<<512, 256, 0, stream>>>(Wfx, Wxt + 1 * HX, X_DIM, H_DIM);
  k_cvt_wT<<<512, 256, 0, stream>>>(Wcx, Wxt + 2 * HX, X_DIM, H_DIM);
  k_cvt_wT<<<512, 256, 0, stream>>>(Wox, Wxt + 3 * HX, X_DIM, H_DIM);
  k_cvt_wT<<<1024, 256, 0, stream>>>(Wih, Wht + 0 * HH, H_DIM, H_DIM);
  k_cvt_wT<<<1024, 256, 0, stream>>>(Wfh, Wht + 1 * HH, H_DIM, H_DIM);
  k_cvt_wT<<<1024, 256, 0, stream>>>(Wch, Wht + 2 * HH, H_DIM, H_DIM);
  k_cvt_wT<<<1024, 256, 0, stream>>>(Woh, Wht + 3 * HH, H_DIM, H_DIM);
  k_cvt_wT<<<1024, 256, 0, stream>>>(Wic, Wct + 0 * HH, H_DIM, H_DIM);
  k_cvt_wT<<<1024, 256, 0, stream>>>(Wfc, Wct + 1 * HH, H_DIM, H_DIM);
  k_cvt_wT<<<1024, 256, 0, stream>>>(Woc, Wct + 2 * HH, H_DIM, H_DIM);

  // fused input projections (32-row M-tiles per block)
  k_proj<<<dim3(T_DIM * B_DIM / 32, (4 * H_DIM) / 512), 256, 0, stream>>>(
      Xbf, Wxt, bi, bfv, bc, bo, P);

  // persistent recurrence: 64 blocks (one per 16 H-columns)
  const size_t smem = (size_t)2 * 64 * LDH * 2 + (size_t)6 * 64 * 16 * 4;
  k_lstm<<<H_DIM / 16, 256, smem, stream>>>(P, Wht, Wct, C32, Hbf, Cbf,
                                            (unsigned*)ws, (float*)d_out);
}
